// Attention_5970004542145
// MI455X (gfx1250) — compile-verified, matrix operand order checked by reading, not run on hardware
//
#include <hip/hip_runtime.h>
#include <hip/hip_bf16.h>

typedef __attribute__((ext_vector_type(16))) _Float16 v16h;
typedef __attribute__((ext_vector_type(8)))  float    v8f;

#define B     64
#define T     2048
#define DRNN  1024
#define DEMB  512
#define DATT  128
#define NF    32
#define KS    31
#define PADW  15
#define TCHUNK 128
#define NCHUNK (T / TCHUNK)   // 16

// Branch-free tanh: 1 - 2/(e^{2x}+1). v_exp_f32 is 2^x, so scale by 2*log2(e).
// Saturates to +/-1 at the extremes (exp2 -> inf/0), no exec divergence.
__device__ __forceinline__ float tanh_fast(float x) {
  float e = __builtin_amdgcn_exp2f(x * 2.8853900817779268f);   // e^{2x}
  return 1.0f - 2.0f * __builtin_amdgcn_rcpf(e + 1.0f);
}
// Branch-free exp for the softmax (inputs are <= 0).
__device__ __forceinline__ float exp_fast(float x) {
  return __builtin_amdgcn_exp2f(x * 1.4426950408889634f);
}

// ---------------------------------------------------------------------------
// k_pq: pq[b][a] = sum_r h[b][r] * Wq[a][r]   (tiny GEMV, pq -> d_ws)
// ---------------------------------------------------------------------------
__global__ __launch_bounds__(256)
void k_pq(const float* __restrict__ h, const float* __restrict__ Wq,
          float* __restrict__ pq) {
  __shared__ float h_lds[DRNN];
  __shared__ float red[256];
  const int b = blockIdx.x, tid = threadIdx.x;
  for (int i = tid; i < DRNN; i += 256) h_lds[i] = h[(size_t)b * DRNN + i];
  __syncthreads();
  const int a    = tid & (DATT - 1);
  const int half = tid >> 7;                     // 0..1, splits the 1024-dot
  const float* wr = Wq + (size_t)a * DRNN + half * (DRNN / 2);
  float acc = 0.f;
  #pragma unroll 8
  for (int r = 0; r < DRNN / 2; ++r) acc += wr[r] * h_lds[half * (DRNN / 2) + r];
  red[tid] = acc;
  __syncthreads();
  if (tid < DATT) pq[b * DATT + tid] = red[tid] + red[tid + DATT];
}

// ---------------------------------------------------------------------------
// k_energy: fused conv1d -> WMMA location-dense -> tanh -> Wv dot.
// One block = one (t-chunk of 128, batch b). 8 waves, each owns a 16-t tile.
// ---------------------------------------------------------------------------
__global__ __launch_bounds__(256)
void k_energy(const float* __restrict__ awcat, const float* __restrict__ convw,
              const float* __restrict__ Wd,    const float* __restrict__ pq,
              const float* __restrict__ Wv,    const float* __restrict__ pmem,
              float* __restrict__ energies) {
  __shared__ float awp[2 * (TCHUNK + KS - 1)];   // padded attn-weights slice
  __shared__ float wd_lds[DATT * NF];            // Wd, row-major [a][f]
  __shared__ float loc_lds[TCHUNK * NF];         // conv output, [t_local][f]
  __shared__ float pq_lds[DATT];
  __shared__ float wv_lds[DATT];

  const int tc  = blockIdx.x;                    // 0..15
  const int b   = blockIdx.y;
  const int tid = threadIdx.x;                   // 0..255
  const int t0  = tc * TCHUNK;

  // ---- stage operands into LDS -------------------------------------------
  for (int i = tid; i < 2 * (TCHUNK + KS - 1); i += 256) {
    int c = i / (TCHUNK + KS - 1);
    int j = i - c * (TCHUNK + KS - 1);
    int t = t0 + j - PADW;
    awp[i] = (t >= 0 && t < T) ? awcat[((size_t)b * 2 + c) * T + t] : 0.f;
  }
  for (int i = tid; i < DATT * NF; i += 256) wd_lds[i] = Wd[i];
  if (tid < DATT) { pq_lds[tid] = pq[b * DATT + tid]; wv_lds[tid] = Wv[tid]; }
  __syncthreads();

  // ---- conv1d(2 -> 32, k=31, same): loc[t][f] ----------------------------
  // Per-thread filter taps cached in registers (contiguous 62-float global
  // read per lane, L2-resident); awp reads are same-address LDS broadcasts.
  {
    const int f = tid & 31, tl0 = tid >> 5;
    float cwreg[2 * KS];
    #pragma unroll
    for (int i = 0; i < 2 * KS; ++i) cwreg[i] = convw[f * (2 * KS) + i];
    for (int p = 0; p < 16; ++p) {
      int tl = tl0 + 8 * p;
      float acc = 0.f;
      #pragma unroll
      for (int c = 0; c < 2; ++c)
        #pragma unroll
        for (int k = 0; k < KS; ++k)
          acc += awp[c * (TCHUNK + KS - 1) + tl + k] * cwreg[c * KS + k];
      loc_lds[tl * NF + f] = acc;
    }
  }

  // ---- per-lane B fragments:  B[K=f][N=a] = Wd[a][f]  (8 a-tiles) --------
  // 16-bit fragment layout (ISA 7.12.2): element i <-> K = (i<8?i:i+8)+8*half
  const int lane = tid & 31, half = lane >> 4, nn = lane & 15;
  v16h bfrag[8];
  #pragma unroll
  for (int j = 0; j < 8; ++j) {
    #pragma unroll
    for (int i = 0; i < 16; ++i) {
      int K = (i < 8 ? i : i + 8) + 8 * half;
      bfrag[j][i] = (_Float16)wd_lds[(j * 16 + nn) * NF + K];
    }
  }
  __syncthreads();

  // ---- WMMA: paw tile = loc^T(16x32) @ Wd^T(32x16), fused epilogue -------
  const int w  = tid >> 5;        // wave id 0..7
  const int tb = w * 16;          // this wave's local t base
  v16h afrag;                     // A[M=t][K=f], M = lane%16
  #pragma unroll
  for (int i = 0; i < 16; ++i) {
    int K = (i < 8 ? i : i + 8) + 8 * half;
    afrag[i] = (_Float16)loc_lds[(tb + nn) * NF + K];
  }
  const float* pm = pmem + ((size_t)b * T + t0 + tb) * DATT;
  float e_acc[8];
  #pragma unroll
  for (int r = 0; r < 8; ++r) e_acc[r] = 0.f;

  #pragma unroll
  for (int j = 0; j < 8; ++j) {
    v8f c = {};
    c = __builtin_amdgcn_wmma_f32_16x16x32_f16(false, afrag, false, bfrag[j],
                                               (short)0, c, false, false);
    const int   a_col = j * 16 + nn;
    const float pqv   = pq_lds[a_col];
    const float wvv   = wv_lds[a_col];
    // C layout: VGPR r, M = r + 8*half, N = lane%16
    #pragma unroll
    for (int r = 0; r < 8; ++r) {
      int   m = r + 8 * half;
      float x = c[r] + pqv + pm[(size_t)m * DATT + a_col];
      e_acc[r] += wvv * tanh_fast(x);
    }
  }
  // reduce over the 16 a-columns held across each 16-lane half
  #pragma unroll
  for (int r = 0; r < 8; ++r) {
    #pragma unroll
    for (int off = 1; off <= 8; off <<= 1)
      e_acc[r] += __shfl_xor(e_acc[r], off, 32);
  }
  if (nn == 0) {
    #pragma unroll
    for (int r = 0; r < 8; ++r) {
      int trow = t0 + tb + r + 8 * half;
      energies[(size_t)b * T + trow] = e_acc[r];
    }
  }
}

// ---------------------------------------------------------------------------
// k_softmax: masked softmax over T, in place in d_out's weights region.
// ---------------------------------------------------------------------------
__global__ __launch_bounds__(256)
void k_softmax(const unsigned char* __restrict__ mask,
               float* __restrict__ wbuf) {
  __shared__ float red[256];
  const int b = blockIdx.x, tid = threadIdx.x;
  float v[8]; int msk[8];
  float mx = -__builtin_inff();
  #pragma unroll
  for (int i = 0; i < 8; ++i) {
    int t = tid + 256 * i;
    msk[i] = mask[(size_t)b * T + t];
    v[i] = msk[i] ? -__builtin_inff() : wbuf[(size_t)b * T + t];
    mx = fmaxf(mx, v[i]);
  }
  red[tid] = mx; __syncthreads();
  for (int s = 128; s > 0; s >>= 1) {
    if (tid < s) red[tid] = fmaxf(red[tid], red[tid + s]);
    __syncthreads();
  }
  mx = red[0]; __syncthreads();
  float sum = 0.f;
  #pragma unroll
  for (int i = 0; i < 8; ++i) {
    v[i] = msk[i] ? 0.f : exp_fast(v[i] - mx);
    sum += v[i];
  }
  red[tid] = sum; __syncthreads();
  for (int s = 128; s > 0; s >>= 1) {
    if (tid < s) red[tid] += red[tid + s];
    __syncthreads();
  }
  const float inv = __builtin_amdgcn_rcpf(red[0]);
  #pragma unroll
  for (int i = 0; i < 8; ++i)
    wbuf[(size_t)b * T + tid + 256 * i] = v[i] * inv;
}

// ---------------------------------------------------------------------------
// k_context: ctx[b][d] = sum_t w[b][t] * mem[b][t][d].
// Grid (4, B): each block owns 128 d's (32 float4 per row -> b128 loads),
// 8-way t-slicing across the 8 waves, LDS reduce at the end.
// ---------------------------------------------------------------------------
__global__ __launch_bounds__(256)
void k_context(const float* __restrict__ wbuf, const float* __restrict__ mem,
               float* __restrict__ ctx) {
  __shared__ float4 red[8 * 32];
  const int s   = blockIdx.x;               // d-range 0..3 (128 floats each)
  const int b   = blockIdx.y;
  const int tid = threadIdx.x;
  const int dl  = tid & 31, ts = tid >> 5;  // lane's float4, 8 t-slices
  const float4* m4 = (const float4*)(mem + (size_t)b * T * DEMB) + s * 32 + dl;
  const float*  wp = wbuf + (size_t)b * T;
  float4 acc = make_float4(0.f, 0.f, 0.f, 0.f);
  for (int t = ts; t < T; t += 8) {
    float  wv = wp[t];
    float4 v  = m4[(size_t)t * (DEMB / 4)];
    acc.x += wv * v.x; acc.y += wv * v.y;
    acc.z += wv * v.z; acc.w += wv * v.w;
  }
  red[ts * 32 + dl] = acc;
  __syncthreads();
  if (tid < 32) {
    float4 r = red[tid];
    #pragma unroll
    for (int i = 1; i < 8; ++i) {
      float4 q = red[i * 32 + tid];
      r.x += q.x; r.y += q.y; r.z += q.z; r.w += q.w;
    }
    ((float4*)(ctx + (size_t)b * DEMB + s * 128))[tid] = r;
  }
}

// ---------------------------------------------------------------------------
extern "C" void kernel_launch(void* const* d_in, const int* in_sizes, int n_in,
                              void* d_out, int out_size, void* d_ws, size_t ws_size,
                              hipStream_t stream) {
  (void)in_sizes; (void)n_in; (void)out_size; (void)ws_size;
  const float* h     = (const float*)d_in[0];  // [B,1024]
  const float* mem   = (const float*)d_in[1];  // [B,T,512]
  const float* pmem  = (const float*)d_in[2];  // [B,T,128]
  const float* awcat = (const float*)d_in[3];  // [B,2,T]
  const unsigned char* mask = (const unsigned char*)d_in[4];  // [B,T] bool
  const float* Wq    = (const float*)d_in[5];  // [128,1024]
  const float* Wv    = (const float*)d_in[6];  // [1,128]
  const float* convw = (const float*)d_in[7];  // [32,2,31]
  const float* Wd    = (const float*)d_in[8];  // [128,32]

  float* out  = (float*)d_out;
  float* ctx  = out;                 // [B,512]  output 0
  float* wbuf = out + B * DEMB;      // [B,T]    output 1 (also energy scratch)
  float* pq   = (float*)d_ws;        // [B,128]  32 KB scratch

  k_pq     <<<dim3(B),          dim3(256), 0, stream>>>(h, Wq, pq);
  k_energy <<<dim3(NCHUNK, B),  dim3(256), 0, stream>>>(awcat, convw, Wd, pq, Wv, pmem, wbuf);
  k_softmax<<<dim3(B),          dim3(256), 0, stream>>>(mask, wbuf);
  k_context<<<dim3(4, B),       dim3(256), 0, stream>>>(wbuf, mem, ctx);
}
